// DeepSeekBlock_11922829213942
// MI455X (gfx1250) — matrix operand order repaired and checked by reading, not compiled, not run on hardware
//
#include <hip/hip_runtime.h>

// ---------------------------------------------------------------------------
// DeepSeek-style block for MI455X (gfx1250, wave32, WMMA + TDM).
// Dominant op: dense expert einsum (1.1 TFLOP) -> v_wmma_f32_16x16x32_bf16.
// Weights/activations pre-converted to bf16 (weights pre-transposed) so the
// GEMM hot loops do zero conversion; tiles staged to LDS by the Tensor Data
// Mover (tensor_load_to_lds + s_wait_tensorcnt), padded 4 DWORDs / 32 DWORDs
// so WMMA fragment ds_load_b128s are 16B-aligned and conflict-free.
// ---------------------------------------------------------------------------

#define USE_TDM 1

#define B_TOK   8192
#define F_DIM   2048
#define E_NUM   16
#define U_DIM   2048
#define DM      2048
#define DEPTH   128

#define BK      64              // K-chunk per stage (2 wmma k-steps)
#define LSTRIDE 72              // halves per LDS row = 64 + 4-DWORD TDM pad

typedef __bf16 v8bf  __attribute__((ext_vector_type(8)));
typedef __bf16 v16bf __attribute__((ext_vector_type(16)));
typedef float  v8f   __attribute__((ext_vector_type(8)));
typedef unsigned int u32x4 __attribute__((ext_vector_type(4)));
typedef int          i32x8 __attribute__((ext_vector_type(8)));
typedef int          i32x4 __attribute__((ext_vector_type(4)));

__device__ __forceinline__ unsigned short bf_round(float f) {
    unsigned u = __builtin_bit_cast(unsigned, f);
    u += 0x7FFFu + ((u >> 16) & 1u);            // round-to-nearest-even
    return (unsigned short)(u >> 16);
}
__device__ __forceinline__ __bf16 f2bf(float f) {
    unsigned short s = bf_round(f);
    return __builtin_bit_cast(__bf16, s);
}
__device__ __forceinline__ float bf2f(__bf16 b) {
    unsigned u = (unsigned)__builtin_bit_cast(unsigned short, b) << 16;
    return __builtin_bit_cast(float, u);
}

// Build a 16-half WMMA fragment from two 16B-aligned LDS chunks.
__device__ __forceinline__ v16bf ld_frag(const __bf16* p0, const __bf16* p1) {
    v8bf a = *(const v8bf*)p0;
    v8bf b = *(const v8bf*)p1;
    v16bf r;
#pragma unroll
    for (int i = 0; i < 8; ++i) { r[i] = a[i]; r[i + 8] = b[i]; }
    return r;
}

#if USE_TDM
// 2D bf16 tile load, global -> LDS, via Tensor Data Mover.
// D# per CDNA5 ISA.  data_size=1 (2B),
// pad_enable: +4 DWORDs every 32 DWORDs -> LDS row stride 72 halves.
// amdgpu-toolchain (clang-23) 6-arg builtin form.
__device__ __forceinline__ void tdm_load_2d(
    unsigned lds_off, const void* gaddr,
    unsigned tensor_d0, unsigned tensor_d1, unsigned stride_d0,
    unsigned tile_d0, unsigned tile_d1)
{
    unsigned long long ga = (unsigned long long)gaddr;
    u32x4 g0;
    g0[0] = 1u;                                      // count=1, user mode
    g0[1] = lds_off;                                 // LDS byte address
    g0[2] = (unsigned)ga;                            // global addr [31:0]
    g0[3] = (unsigned)(ga >> 32) | (2u << 30);       // addr [56:32] | type=2
    unsigned w[8];
    w[0] = (1u << 16)                                // data_size = 2 bytes
         | (1u << 20)                                // pad_enable
         | (4u << 22)                                // pad_interval: 32 DWORDs
         | (3u << 25);                               // pad_amount: 4 DWORDs
    w[1] = tensor_d0 << 16;                          // tensor_dim0 [15:0]
    w[2] = (tensor_d0 >> 16) | (tensor_d1 << 16);    // dim0 hi | dim1 lo
    w[3] = (tensor_d1 >> 16) | (tile_d0 << 16);      // dim1 hi | tile_dim0
    w[4] = tile_d1 & 0xFFFFu;                        // tile_dim1 (tile_dim2=0)
    w[5] = stride_d0;                                // tensor_dim0_stride lo
    w[6] = 0u;
    w[7] = 0u;
    i32x8 g1;
#pragma unroll
    for (int i = 0; i < 8; ++i) g1[i] = (int)w[i];
    i32x4 z4 = {0, 0, 0, 0};
    i32x8 z8 = {0, 0, 0, 0, 0, 0, 0, 0};
    __builtin_amdgcn_tensor_load_to_lds(g0, g1, z4, z4, z8, 0);
}
#endif

// ---------------------------------------------------------------------------
// One-time conversion kernels (stream once; ~20us of BW total).
// ---------------------------------------------------------------------------
__global__ __launch_bounds__(256) void f32_to_bf16_kernel(
    const float* __restrict__ src, __bf16* __restrict__ dst)
{
    int i = blockIdx.x * 256 + threadIdx.x;          // group of 4 elements
    float4 f = ((const float4*)src)[i];
    uint2 o;
    o.x = (unsigned)bf_round(f.x) | ((unsigned)bf_round(f.y) << 16);
    o.y = (unsigned)bf_round(f.z) | ((unsigned)bf_round(f.w) << 16);
    ((uint2*)dst)[i] = o;
}

// dst[c][r] (bf16) = src[r][c] (fp32); R,C multiples of 32; blockIdx.z = matrix
__global__ __launch_bounds__(256) void transpose_bf16_kernel(
    const float* __restrict__ src, __bf16* __restrict__ dst, int R, int C)
{
    __shared__ __bf16 tile[32 * 33];
    size_t mat = (size_t)blockIdx.z * (size_t)R * C;
    src += mat; dst += mat;
    int r0 = blockIdx.x * 32, c0 = blockIdx.y * 32, t = threadIdx.x;
#pragma unroll
    for (int i = 0; i < 4; ++i) {
        int el = i * 256 + t, rr = el >> 5, cc = el & 31;
        tile[rr * 33 + cc] = f2bf(src[(size_t)(r0 + rr) * C + c0 + cc]);
    }
    __syncthreads();
#pragma unroll
    for (int i = 0; i < 4; ++i) {
        int el = i * 256 + t, cc = el >> 5, rr = el & 31;
        dst[(size_t)(c0 + cc) * R + r0 + rr] = tile[rr * 33 + cc];
    }
}

// ---------------------------------------------------------------------------
// Router: gate[b,e] = softmax(x@rw + rb)[e] if e in top-2 else 0.  1 wave/token.
// ---------------------------------------------------------------------------
__global__ __launch_bounds__(256) void router_kernel(
    const float* __restrict__ x, const float* __restrict__ rw,
    const float* __restrict__ rb, float* __restrict__ gate)
{
    int wave = threadIdx.x >> 5, lane = threadIdx.x & 31;
    int b = blockIdx.x * 8 + wave;
    const float* xb = x + (size_t)b * F_DIM;

    float acc[E_NUM];
#pragma unroll
    for (int e = 0; e < E_NUM; ++e) acc[e] = 0.f;
    for (int f = lane; f < F_DIM; f += 32) {
        float xv = xb[f];
        const float4* wr = (const float4*)(rw + (size_t)f * E_NUM);
#pragma unroll
        for (int q = 0; q < 4; ++q) {
            float4 w4 = wr[q];
            acc[q * 4 + 0] += xv * w4.x;  acc[q * 4 + 1] += xv * w4.y;
            acc[q * 4 + 2] += xv * w4.z;  acc[q * 4 + 3] += xv * w4.w;
        }
    }
#pragma unroll
    for (int e = 0; e < E_NUM; ++e) {
#pragma unroll
        for (int off = 16; off; off >>= 1) acc[e] += __shfl_xor(acc[e], off, 32);
        acc[e] += rb[e];
    }
    float mx = acc[0];
#pragma unroll
    for (int e = 1; e < E_NUM; ++e) mx = fmaxf(mx, acc[e]);
    float p[E_NUM], sum = 0.f;
#pragma unroll
    for (int e = 0; e < E_NUM; ++e) { p[e] = __expf(acc[e] - mx); sum += p[e]; }
    float inv = 1.f / sum;
#pragma unroll
    for (int e = 0; e < E_NUM; ++e) p[e] *= inv;
    float b1 = -1.f; int i1 = 0;
#pragma unroll
    for (int e = 0; e < E_NUM; ++e) if (p[e] > b1) { b1 = p[e]; i1 = e; }
    float b2 = -1.f; int i2 = 0;
#pragma unroll
    for (int e = 0; e < E_NUM; ++e) if (e != i1 && p[e] > b2) { b2 = p[e]; i2 = e; }
    if (lane == 0) {
#pragma unroll
        for (int e = 0; e < E_NUM; ++e)
            gate[(size_t)b * E_NUM + e] = (e == i1 || e == i2) ? p[e] : 0.f;
    }
}

// ---------------------------------------------------------------------------
// MoE dense-expert WMMA GEMM with gated-ReLU reduction (bf16 in, bf16 out).
// Tile 32(M) x 256(N), 8 waves (2x4 of 16x64); expert-outer / K-inner (BK=64)
// with block-uniform expert skip.  A = x_bf [B,F], B = ewT [E*U, F] (K-contig).
// ---------------------------------------------------------------------------
__global__ __launch_bounds__(256) void moe_kernel(
    const __bf16* __restrict__ xb, const float* __restrict__ gate,
    const __bf16* __restrict__ ewT, const float* __restrict__ eb,
    __bf16* __restrict__ moe)
{
    __shared__ __bf16 Xs[32 * LSTRIDE];     //  4.5 KB
    __shared__ __bf16 Ws[256 * LSTRIDE];    // 36.0 KB
    __shared__ float  Gs[32 * E_NUM];
    __shared__ float  EBs[E_NUM * 256];

    int tid = threadIdx.x, wave = tid >> 5, lane = tid & 31;
    int m0 = blockIdx.x * 32, n0 = blockIdx.y * 256;

    for (int i = tid; i < 32 * E_NUM; i += 256)
        Gs[i] = gate[(size_t)m0 * E_NUM + i];
    for (int i = tid; i < E_NUM * 256; i += 256)
        EBs[i] = eb[(size_t)(i >> 8) * U_DIM + n0 + (i & 255)];
    __syncthreads();

    int wm = wave >> 2, wn = wave & 3;
    int mBase = wm * 16, nBase = wn * 64;
    int hi = lane >> 4, lo16 = lane & 15;

    float outacc[4][8];
#pragma unroll
    for (int j = 0; j < 4; ++j)
#pragma unroll
        for (int r = 0; r < 8; ++r) outacc[j][r] = 0.f;

    for (int e = 0; e < E_NUM; ++e) {
        float gv = Gs[lane * E_NUM + e];
#pragma unroll
        for (int off = 16; off; off >>= 1) gv = fmaxf(gv, __shfl_xor(gv, off, 32));
        if (gv <= 0.f) continue;             // uniform across block

        v8f acc[4];
#pragma unroll
        for (int j = 0; j < 4; ++j)
#pragma unroll
            for (int r = 0; r < 8; ++r) acc[j][r] = 0.f;

        for (int k0 = 0; k0 < F_DIM; k0 += BK) {
#if USE_TDM
            if (wave == 0) {
                tdm_load_2d((unsigned)(size_t)&Xs[0],
                            xb + (size_t)m0 * F_DIM + k0,
                            F_DIM, B_TOK, F_DIM, BK, 32);
                tdm_load_2d((unsigned)(size_t)&Ws[0],
                            ewT + ((size_t)e * U_DIM + n0) * F_DIM + k0,
                            F_DIM, E_NUM * U_DIM, F_DIM, BK, 256);
                __builtin_amdgcn_s_wait_tensorcnt(0);
            }
#else
            {
                int r = tid >> 3, c = (tid & 7) * 8;
                *(uint4*)&Xs[r * LSTRIDE + c] =
                    *(const uint4*)(xb + (size_t)(m0 + r) * F_DIM + k0 + c);
                const __bf16* wp = ewT + ((size_t)e * U_DIM + n0 + tid) * F_DIM + k0;
#pragma unroll
                for (int q = 0; q < 4; ++q)
                    *(uint4*)&Ws[tid * LSTRIDE + q * 8] = *(const uint4*)(wp + q * 8);
            }
#endif
            __syncthreads();
#pragma unroll
            for (int ks = 0; ks < BK; ks += 32) {
                const __bf16* pa = &Xs[(mBase + lo16) * LSTRIDE + ks + hi * 8];
                v16bf afr = ld_frag(pa, pa + 16);
#pragma unroll
                for (int j = 0; j < 4; ++j) {
                    const __bf16* pb =
                        &Ws[(nBase + j * 16 + lo16) * LSTRIDE + ks + hi * 8];
                    v16bf bfr = ld_frag(pb, pb + 16);
                    acc[j] = __builtin_amdgcn_wmma_f32_16x16x32_bf16(
                        false, afr, false, bfr, (short)0, acc[j], false, false);
                }
            }
            __syncthreads();
        }
        float g[8];
#pragma unroll
        for (int r = 0; r < 8; ++r) g[r] = Gs[(mBase + r + hi * 8) * E_NUM + e];
#pragma unroll
        for (int j = 0; j < 4; ++j) {
            float bia = EBs[e * 256 + nBase + j * 16 + lo16];
#pragma unroll
            for (int r = 0; r < 8; ++r)
                outacc[j][r] += g[r] * fmaxf(acc[j][r] + bia, 0.f);
        }
    }
#pragma unroll
    for (int j = 0; j < 4; ++j)
#pragma unroll
        for (int r = 0; r < 8; ++r) {
            int m = m0 + mBase + r + hi * 8;
            int n = n0 + nBase + j * 16 + lo16;
            moe[(size_t)m * U_DIM + n] = f2bf(outacc[j][r]);
        }
}

// ---------------------------------------------------------------------------
// Generic WMMA GEMM: C[M,N] = A[M,2048](bf16) @ WT[N,2048]^T (bf16) + bias.
// Writes fp32 (Cf) and/or bf16 (Cb).
// ---------------------------------------------------------------------------
__global__ __launch_bounds__(256) void gemm2048_kernel(
    const __bf16* __restrict__ A, const __bf16* __restrict__ WT,
    const float* __restrict__ bias, float* __restrict__ Cf,
    __bf16* __restrict__ Cb, int N)
{
    __shared__ __bf16 Xs[32 * LSTRIDE];
    __shared__ __bf16 Ws[256 * LSTRIDE];
    __shared__ float  Bs[256];

    int tid = threadIdx.x, wave = tid >> 5, lane = tid & 31;
    int m0 = blockIdx.x * 32, n0 = blockIdx.y * 256;
    Bs[tid] = bias[n0 + tid];

    int wm = wave >> 2, wn = wave & 3;
    int mBase = wm * 16, nBase = wn * 64;
    int hi = lane >> 4, lo16 = lane & 15;

    v8f acc[4];
#pragma unroll
    for (int j = 0; j < 4; ++j)
#pragma unroll
        for (int r = 0; r < 8; ++r) acc[j][r] = 0.f;

    __syncthreads();
    for (int k0 = 0; k0 < 2048; k0 += BK) {
#if USE_TDM
        if (wave == 0) {
            tdm_load_2d((unsigned)(size_t)&Xs[0],
                        A + (size_t)m0 * 2048 + k0,
                        2048, B_TOK, 2048, BK, 32);
            tdm_load_2d((unsigned)(size_t)&Ws[0],
                        WT + (size_t)n0 * 2048 + k0,
                        2048, (unsigned)N, 2048, BK, 256);
            __builtin_amdgcn_s_wait_tensorcnt(0);
        }
#else
        {
            int r = tid >> 3, c = (tid & 7) * 8;
            *(uint4*)&Xs[r * LSTRIDE + c] =
                *(const uint4*)(A + (size_t)(m0 + r) * 2048 + k0 + c);
            const __bf16* wp = WT + (size_t)(n0 + tid) * 2048 + k0;
#pragma unroll
            for (int q = 0; q < 4; ++q)
                *(uint4*)&Ws[tid * LSTRIDE + q * 8] = *(const uint4*)(wp + q * 8);
        }
#endif
        __syncthreads();
#pragma unroll
        for (int ks = 0; ks < BK; ks += 32) {
            const __bf16* pa = &Xs[(mBase + lo16) * LSTRIDE + ks + hi * 8];
            v16bf afr = ld_frag(pa, pa + 16);
#pragma unroll
            for (int j = 0; j < 4; ++j) {
                const __bf16* pb =
                    &Ws[(nBase + j * 16 + lo16) * LSTRIDE + ks + hi * 8];
                v16bf bfr = ld_frag(pb, pb + 16);
                acc[j] = __builtin_amdgcn_wmma_f32_16x16x32_bf16(
                    false, afr, false, bfr, (short)0, acc[j], false, false);
            }
        }
        __syncthreads();
    }
#pragma unroll
    for (int j = 0; j < 4; ++j) {
        int n = nBase + j * 16 + lo16;
#pragma unroll
        for (int r = 0; r < 8; ++r) {
            int m = m0 + mBase + r + hi * 8;
            float v = acc[j][r] + Bs[n];
            size_t idx = (size_t)m * N + n0 + n;
            if (Cf) Cf[idx] = v;
            if (Cb) Cb[idx] = f2bf(v);
        }
    }
}

// ---------------------------------------------------------------------------
// Per-token head-softmax attention: y = softmax_h(q.k/sqrt(D)) * v (bf16 io).
// One wave per token, 2 lanes per head.  y may alias q (q consumed first).
// ---------------------------------------------------------------------------
__global__ __launch_bounds__(256) void attn_kernel(
    const __bf16* __restrict__ q, const __bf16* __restrict__ k,
    const __bf16* __restrict__ v, __bf16* __restrict__ y)
{
    int wave = threadIdx.x >> 5, lane = threadIdx.x & 31;
    int b = blockIdx.x * 8 + wave;
    int h = lane >> 1, part = lane & 1;
    int base = h * DEPTH + part * 64;
    const __bf16* qb = q + (size_t)b * DM;
    const __bf16* kb = k + (size_t)b * DM;
    const __bf16* vb = v + (size_t)b * DM;

    float s = 0.f;
#pragma unroll 8
    for (int j = 0; j < 64; ++j) s += bf2f(qb[base + j]) * bf2f(kb[base + j]);
    s += __shfl_xor(s, 1, 32);
    s *= 0.08838834764831845f;               // 1/sqrt(128)

    float mx = s;
#pragma unroll
    for (int off = 2; off < 32; off <<= 1) mx = fmaxf(mx, __shfl_xor(mx, off, 32));
    float ex = __expf(s - mx);
    float sum = ex;
#pragma unroll
    for (int off = 2; off < 32; off <<= 1) sum += __shfl_xor(sum, off, 32);
    float attn = ex / sum;

    __bf16* yb = y + (size_t)b * DM;
#pragma unroll 8
    for (int j = 0; j < 64; ++j) yb[base + j] = f2bf(attn * bf2f(vb[base + j]));
}

// ---------------------------------------------------------------------------
extern "C" void kernel_launch(void* const* d_in, const int* in_sizes, int n_in,
                              void* d_out, int out_size, void* d_ws, size_t ws_size,
                              hipStream_t stream)
{
    (void)in_sizes; (void)n_in; (void)out_size; (void)ws_size;
    const float* x        = (const float*)d_in[0];
    const float* router_w = (const float*)d_in[1];
    const float* router_b = (const float*)d_in[2];
    const float* expert_w = (const float*)d_in[3];
    const float* expert_b = (const float*)d_in[4];
    const float* wq = (const float*)d_in[5];  const float* bq = (const float*)d_in[6];
    const float* wk = (const float*)d_in[7];  const float* bk = (const float*)d_in[8];
    const float* wv = (const float*)d_in[9];  const float* bv = (const float*)d_in[10];
    const float* wo = (const float*)d_in[11]; const float* bo = (const float*)d_in[12];
    float* out = (float*)d_out;

    char* p = (char*)d_ws;
    auto alloc = [&](size_t bytes) {
        char* r = p; p += (bytes + 255) & ~(size_t)255; return r;
    };
    const size_t BF = (size_t)B_TOK * F_DIM;       // 16.7M
    const size_t KN = (size_t)F_DIM * DM;          // 4.2M
    __bf16* x_bf  = (__bf16*)alloc(BF * 2);
    __bf16* ewT   = (__bf16*)alloc((size_t)E_NUM * KN * 2);
    __bf16* wqT   = (__bf16*)alloc(KN * 2);
    __bf16* wkT   = (__bf16*)alloc(KN * 2);
    __bf16* wvT   = (__bf16*)alloc(KN * 2);
    __bf16* woT   = (__bf16*)alloc(KN * 2);
    float*  gatep = (float*)alloc((size_t)B_TOK * E_NUM * 4);
    __bf16* moe   = (__bf16*)alloc(BF * 2);
    __bf16* qb    = (__bf16*)alloc(BF * 2);
    __bf16* kb    = (__bf16*)alloc(BF * 2);
    __bf16* vb    = (__bf16*)alloc(BF * 2);

    // One-time conversions / transposes.
    f32_to_bf16_kernel<<<(int)(BF / 4 / 256), 256, 0, stream>>>(x, x_bf);
    transpose_bf16_kernel<<<dim3(F_DIM / 32, U_DIM / 32, E_NUM), 256, 0, stream>>>(
        expert_w, ewT, F_DIM, U_DIM);
    transpose_bf16_kernel<<<dim3(F_DIM / 32, DM / 32, 1), 256, 0, stream>>>(wq, wqT, F_DIM, DM);
    transpose_bf16_kernel<<<dim3(F_DIM / 32, DM / 32, 1), 256, 0, stream>>>(wk, wkT, F_DIM, DM);
    transpose_bf16_kernel<<<dim3(F_DIM / 32, DM / 32, 1), 256, 0, stream>>>(wv, wvT, F_DIM, DM);
    transpose_bf16_kernel<<<dim3(F_DIM / 32, DM / 32, 1), 256, 0, stream>>>(wo, woT, F_DIM, DM);

    // Pipeline.
    router_kernel<<<B_TOK / 8, 256, 0, stream>>>(x, router_w, router_b, gatep);
    moe_kernel<<<dim3(B_TOK / 32, U_DIM / 256), 256, 0, stream>>>(
        x_bf, gatep, ewT, expert_b, moe);
    gemm2048_kernel<<<dim3(B_TOK / 32, DM / 256), 256, 0, stream>>>(
        moe, wqT, bq, nullptr, qb, DM);
    gemm2048_kernel<<<dim3(B_TOK / 32, DM / 256), 256, 0, stream>>>(
        moe, wkT, bk, nullptr, kb, DM);
    gemm2048_kernel<<<dim3(B_TOK / 32, DM / 256), 256, 0, stream>>>(
        moe, wvT, bv, nullptr, vb, DM);
    attn_kernel<<<B_TOK / 8, 256, 0, stream>>>(qb, kb, vb, qb);   // y overwrites q
    gemm2048_kernel<<<dim3(B_TOK / 32, DM / 256), 256, 0, stream>>>(
        qb, woT, bo, out, nullptr, DM);
}
